// GraphIoT_71055938945215
// MI455X (gfx1250) — compile-verified, matrix operand order checked by reading, not run on hardware
//
#include <hip/hip_runtime.h>
#include <hip/hip_bf16.h>

// ---------------------------------------------------------------------------
// GIN forward for MI455X (gfx1250, wave32, WMMA).
// Dense 256x256 GEMMs: v_wmma_f32_16x16x32_bf16, B staged in LDS via
// global_load_async_to_lds_b128 (ASYNCcnt), A double-buffered from global.
// Edge aggregation uses f32 global atomics (the memory-bound part).
// ---------------------------------------------------------------------------

#define RANKC   256
#define NNODES  131072
#define LFLOW   128
#define NFLOWS  1024   // NNODES / LFLOW
#define NCLS    32
#define BN_EPS  1e-5f

typedef __attribute__((ext_vector_type(16))) __bf16 v16bf;
typedef __attribute__((ext_vector_type(8)))  __bf16 v8bf;
typedef __attribute__((ext_vector_type(8)))  float  v8f;

typedef __attribute__((address_space(3))) unsigned short lds_us;

// round-to-nearest-even float -> bf16 (as raw u16)
__device__ __forceinline__ unsigned short f2bf(float x) {
    unsigned u = __float_as_uint(x);
    u += 0x7FFFu + ((u >> 16) & 1u);
    return (unsigned short)(u >> 16);
}

// ---------------------------------------------------------------------------
// Repack W[9][256][256] (f32, row-major: W[k][n]) into WMMA-B fragments,
// grouped so each 64-column N-strip is contiguous (32KB) for async staging:
//   Bp[mat][ns(4)][kb(8)][t(4)][lane(32)][e(16)]  (bf16)
// 16-bit B 32x16 layout: lane L (<16) holds col N=L with K in {0..7,16..23}
// (+8 for lanes 16..31), packed pairwise into 8 VGPRs.
// ---------------------------------------------------------------------------
__global__ void repack_w_kernel(const float* __restrict__ W,
                                unsigned short* __restrict__ Bp) {
    size_t i = (size_t)blockIdx.x * blockDim.x + threadIdx.x; // 9*65536
    int mat  = (int)(i >> 16);
    int o    = (int)(i & 65535);
    int e    = o & 15;
    int lane = (o >> 4) & 31;
    int t    = (o >> 9) & 3;
    int kb   = (o >> 11) & 7;
    int ns   = o >> 14;
    int n     = (ns * 4 + t) * 16 + (lane & 15);
    int kbase = kb * 32 + ((lane >> 4) ? 8 : 0);
    int k     = kbase + (e < 8 ? e : 8 + e);   // e=8 -> kbase+16 ... e=15 -> kbase+23
    Bp[i] = f2bf(W[(size_t)mat * 65536 + (size_t)k * RANKC + n]);
}

// seq_encoder: feats[n][c] = node_vals[n] * seq_W[c] + seq_b[c]
__global__ void encode_kernel(const float* __restrict__ nv,
                              const float* __restrict__ sw,
                              const float* __restrict__ sb,
                              float* __restrict__ feats) {
    size_t i = (size_t)blockIdx.x * blockDim.x + threadIdx.x;
    int c = (int)(i & (RANKC - 1));
    size_t n = i >> 8;
    feats[i] = nv[n] * sw[c] + sb[c];
}

__global__ void zero_kernel(float* __restrict__ p, int n) {
    int i = blockIdx.x * blockDim.x + threadIdx.x;
    if (i < n) p[i] = 0.f;
}

// Edge scatter: 64 threads per edge, 4 floats per thread (float4 gather,
// 4x global_atomic_add_f32 scatter). agg pre-initialized to feats, so after
// this agg = feats + segment_sum(feats[src], dst).
__global__ void scatter_add_kernel(const int* __restrict__ esrc,
                                   const int* __restrict__ edst,
                                   const float* __restrict__ feats,
                                   float* __restrict__ agg,
                                   long long nE) {
    long long t = (long long)blockIdx.x * blockDim.x + threadIdx.x;
    long long e = t >> 6;
    if (e >= nE) return;
    int d4 = ((int)t & 63) * 4;
    long long s = esrc[e];
    long long d = edst[e];
    const float4 v = *(const float4*)(feats + s * RANKC + d4);
    float* a = agg + d * RANKC + d4;
    atomicAdd(a + 0, v.x);
    atomicAdd(a + 1, v.y);
    atomicAdd(a + 2, v.z);
    atomicAdd(a + 3, v.w);
}

// f32 -> bf16 convert (GEMM input)
__global__ void cvt_bf16_kernel(const float* __restrict__ x,
                                unsigned short* __restrict__ y) {
    size_t i = (size_t)blockIdx.x * blockDim.x + threadIdx.x;
    y[i] = f2bf(x[i]);
}

// ---------------------------------------------------------------------------
// z = relu(h @ W + b), h: [M,256] bf16 row-major, W repacked, z: [M,256] f32.
// Block = 256 threads = 8 waves; wave w covers rows blockIdx.x*128 + w*16,
// cols blockIdx.y*64 .. +63 (4 WMMA n-tiles).
// B strip (32KB) is async-copied into LDS once per block and shared by all
// 8 waves; A fragments are double-buffered from global.
// ---------------------------------------------------------------------------
__global__ void __launch_bounds__(256)
gemm_relu_kernel(const unsigned short* __restrict__ h,
                 const unsigned short* __restrict__ Bmat,
                 const float* __restrict__ bias,
                 float* __restrict__ z) {
    __shared__ unsigned short smem[16384];   // [kb(8)][t(4)][lane(32)][e(16)] = 32KB

    const int lane    = threadIdx.x & 31;
    const int wave    = threadIdx.x >> 5;
    const int mbase   = blockIdx.x * 128 + wave * 16;
    const int nstrip  = blockIdx.y;               // 0..3
    const int halfsel = lane >> 4;                // 0 or 1
    const int arow    = mbase + (lane & 15);      // A-fragment row for this lane

    // --- Stage this block's B strip into LDS with async copies (16B/lane) ---
    const unsigned short* Bstrip = Bmat + (size_t)nstrip * 16384;
#pragma unroll
    for (int pass = 0; pass < 8; ++pass) {
        int idx = pass * 256 + threadIdx.x;               // 16B chunk id (0..2047)
        const unsigned short* g = Bstrip + (size_t)idx * 8;
        lds_us* lp = (lds_us*)&smem[idx * 8];
        asm volatile("global_load_async_to_lds_b128 %0, %1, off"
                     :: "v"(lp), "v"(g) : "memory");
    }
    asm volatile("s_wait_asynccnt 0x0" ::: "memory");
    __syncthreads();

    // --- K loop: 8 steps of 32, A double-buffered, B from LDS -------------
    v8f acc0 = {}, acc1 = {}, acc2 = {}, acc3 = {};
    union AV { v16bf v; v8bf h2[2]; } A[2];
    const unsigned short* abase = h + (size_t)arow * RANKC + halfsel * 8;
    A[0].h2[0] = *(const v8bf*)(abase);        // K = 0..7 (+8 for upper half)
    A[0].h2[1] = *(const v8bf*)(abase + 16);   // K = 16..23 (+8)

#pragma unroll
    for (int kb = 0; kb < 8; ++kb) {
        if (kb < 7) {
            const unsigned short* ap = abase + (kb + 1) * 32;
            A[(kb + 1) & 1].h2[0] = *(const v8bf*)(ap);
            A[(kb + 1) & 1].h2[1] = *(const v8bf*)(ap + 16);
        }
        const unsigned short* bl = &smem[(kb * 4 * 32 + lane) * 16];
        v16bf B0 = *(const v16bf*)(bl);
        v16bf B1 = *(const v16bf*)(bl + 512);
        v16bf B2 = *(const v16bf*)(bl + 1024);
        v16bf B3 = *(const v16bf*)(bl + 1536);
        v16bf Av = A[kb & 1].v;
        acc0 = __builtin_amdgcn_wmma_f32_16x16x32_bf16(false, Av, false, B0,
                                                       (short)0, acc0, false, false);
        acc1 = __builtin_amdgcn_wmma_f32_16x16x32_bf16(false, Av, false, B1,
                                                       (short)0, acc1, false, false);
        acc2 = __builtin_amdgcn_wmma_f32_16x16x32_bf16(false, Av, false, B2,
                                                       (short)0, acc2, false, false);
        acc3 = __builtin_amdgcn_wmma_f32_16x16x32_bf16(false, Av, false, B3,
                                                       (short)0, acc3, false, false);
    }

    // Epilogue: C/D layout -> row = mbase + halfsel*8 + r, col = tile + (lane&15)
    const int rowb = mbase + halfsel * 8;
    const int colb = nstrip * 64 + (lane & 15);
    v8f accs[4] = {acc0, acc1, acc2, acc3};
#pragma unroll
    for (int t = 0; t < 4; ++t) {
        const int col = colb + t * 16;
        const float bc = bias[col];
#pragma unroll
        for (int r = 0; r < 8; ++r) {
            float v = accs[t][r] + bc;
            z[(size_t)(rowb + r) * RANKC + col] = v > 0.f ? v : 0.f;
        }
    }
}

// Per-feature sum / sumsq over all rows (atomics into 512-float slice).
__global__ void bn_stats_kernel(const float* __restrict__ z,
                                float* __restrict__ stats) {
    const int f  = threadIdx.x;
    const int r0 = blockIdx.x * 512;
    float s = 0.f, q = 0.f;
    const float* p = z + (size_t)r0 * RANKC + f;
    for (int r = 0; r < 512; ++r) {
        float v = p[(size_t)r * RANKC];
        s += v;
        q += v * v;
    }
    atomicAdd(&stats[f], s);
    atomicAdd(&stats[RANKC + f], q);
}

// BN apply (+ bf16 convert for next GEMM, + optional f32 feats for readout/agg)
__global__ void bn_apply_kernel(const float* __restrict__ z,
                                const float* __restrict__ stats,
                                const float* __restrict__ gamma,
                                const float* __restrict__ beta,
                                unsigned short* __restrict__ hb,
                                float* __restrict__ feats_out) {
    size_t i = (size_t)blockIdx.x * blockDim.x + threadIdx.x;
    int f = (int)(i & (RANKC - 1));
    const float invN = 1.f / (float)NNODES;
    float mean = stats[f] * invN;
    float var  = stats[RANKC + f] * invN - mean * mean;
    float rs   = rsqrtf(var + BN_EPS);
    float y = (z[i] - mean) * rs * gamma[f] + beta[f];
    hb[i] = f2bf(y);
    if (feats_out) feats_out[i] = y;
}

// Per-flow readout: sum 128 rows -> ro[flow][layer*256 + f]
__global__ void readout_kernel(const float* __restrict__ feats,
                               float* __restrict__ ro, int layer) {
    const int flow = blockIdx.x;
    const int f    = threadIdx.x;
    const float* p = feats + (size_t)flow * LFLOW * RANKC + f;
    float s = 0.f;
    for (int r = 0; r < LFLOW; ++r) s += p[(size_t)r * RANKC];
    ro[(size_t)flow * (3 * RANKC) + layer * RANKC + f] = s;
}

// Classifier: out[1024][32] = ro[1024][768] @ clfW[768][32] + clfB
__global__ void classifier_kernel(const float* __restrict__ ro,
                                  const float* __restrict__ W,
                                  const float* __restrict__ b,
                                  float* __restrict__ out) {
    const int row = blockIdx.x;
    const int c   = threadIdx.x;  // 32
    const float* r = ro + (size_t)row * (3 * RANKC);
    float s = b[c];
    for (int k = 0; k < 3 * RANKC; ++k) s += r[k] * W[(size_t)k * NCLS + c];
    out[(size_t)row * NCLS + c] = s;
}

// ---------------------------------------------------------------------------
extern "C" void kernel_launch(void* const* d_in, const int* in_sizes, int n_in,
                              void* d_out, int out_size, void* d_ws, size_t ws_size,
                              hipStream_t stream) {
    const float* node_vals = (const float*)d_in[0];
    const int*   edge_src  = (const int*)d_in[1];
    const int*   edge_dst  = (const int*)d_in[2];
    const float* seq_W     = (const float*)d_in[3];   // [1,256]
    const float* seq_b     = (const float*)d_in[4];
    const float* mlp_W     = (const float*)d_in[5];   // [3,3,256,256]
    const float* mlp_b     = (const float*)d_in[6];   // [3,3,256]
    const float* bn_gamma  = (const float*)d_in[7];
    const float* bn_beta   = (const float*)d_in[8];
    const float* clf_W     = (const float*)d_in[9];   // [768,32]
    const float* clf_b     = (const float*)d_in[10];
    float* out = (float*)d_out;

    const long long nE = (long long)in_sizes[1];
    const size_t NF = (size_t)NNODES * RANKC;        // 33.55M elems

    // Workspace carve-up
    char* p = (char*)d_ws;
    float* feats = (float*)p;            p += NF * 4;        // 128 MB
    float* agg   = (float*)p;            p += NF * 4;        // 128 MB
    float* zbuf  = (float*)p;            p += NF * 4;        // 128 MB
    unsigned short* hb = (unsigned short*)p;  p += NF * 2;   // 64 MB
    unsigned short* Bp = (unsigned short*)p;  p += 9 * 65536 * 2;
    float* stats = (float*)p;            p += 9 * 512 * 4;
    float* ro    = (float*)p;            p += (size_t)NFLOWS * 3 * RANKC * 4;

    // 1) Repack weights into WMMA-B fragment layout (cheap, every call)
    repack_w_kernel<<<(9 * 65536) / 256, 256, 0, stream>>>(mlp_W, Bp);
    // 2) Zero BN stats accumulators
    zero_kernel<<<(9 * 512 + 255) / 256, 256, 0, stream>>>(stats, 9 * 512);
    // 3) seq_encoder
    encode_kernel<<<(int)(NF / 256), 256, 0, stream>>>(node_vals, seq_W, seq_b, feats);

    for (int layer = 0; layer < 3; ++layer) {
        // agg = feats (GIN eps=0 self term), then += segment_sum over edges
        hipMemcpyAsync(agg, feats, NF * 4, hipMemcpyDeviceToDevice, stream);
        {
            long long threads = nE * 64;
            int blocks = (int)((threads + 255) / 256);
            scatter_add_kernel<<<blocks, 256, 0, stream>>>(edge_src, edge_dst,
                                                           feats, agg, nE);
        }
        cvt_bf16_kernel<<<(int)(NF / 256), 256, 0, stream>>>(agg, hb);

        for (int j = 0; j < 3; ++j) {
            const int s = layer * 3 + j;
            dim3 grid(NNODES / 128, 4);
            gemm_relu_kernel<<<grid, 256, 0, stream>>>(
                hb, Bp + (size_t)s * 65536, mlp_b + (size_t)s * RANKC, zbuf);
            bn_stats_kernel<<<NNODES / 512, 256, 0, stream>>>(zbuf, stats + s * 512);
            bn_apply_kernel<<<(int)(NF / 256), 256, 0, stream>>>(
                zbuf, stats + s * 512,
                bn_gamma + (size_t)s * RANKC, bn_beta + (size_t)s * RANKC,
                hb, (j == 2) ? feats : nullptr);
        }
        readout_kernel<<<NFLOWS, RANKC, 0, stream>>>(feats, ro, layer);
    }

    classifier_kernel<<<NFLOWS, NCLS, 0, stream>>>(ro, clf_W, clf_b, out);
}
